// GATlayer_64175401337197
// MI455X (gfx1250) — compile-verified, matrix operand order checked by reading, not run on hardware
//
#include <hip/hip_runtime.h>
#include <hip/hip_bf16.h>

// GAT layer for MI455X (gfx1250), wave32 + WMMA f32_16x16x32_f16.
// B=512, S=256, D=256. Two WMMA GEMMs + fused softmax; ~400MB HBM traffic.

typedef __attribute__((ext_vector_type(16))) _Float16 v16h;
typedef __attribute__((ext_vector_type(8)))  _Float16 v8h;
typedef __attribute__((ext_vector_type(8)))  float    v8f;

#define Bc 512
#define Sc 256
#define Dc 256
#define NROW (Bc * Sc)          // 131072 tokens
#define NEG_SLOPE 0.1f

// ---------------------------------------------------------------------------
// A-fragment (16x32 f16) from row-major f32: lane L holds row m = L&15,
// K values {klo..klo+7, klo+16..klo+23}, klo = 8*(L>>4).
// base points at element (m, k0+klo) of a 256-wide row.
// ---------------------------------------------------------------------------
__device__ inline v16h frag_from_f32(const float* base) {
    const float4* p = (const float4*)base;
    float4 f0 = p[0];          // k+0..3
    float4 f1 = p[1];          // k+4..7
    float4 f2 = p[4];          // k+16..19
    float4 f3 = p[5];          // k+20..23
    v16h r;
    r[0]=(_Float16)f0.x; r[1]=(_Float16)f0.y; r[2]=(_Float16)f0.z; r[3]=(_Float16)f0.w;
    r[4]=(_Float16)f1.x; r[5]=(_Float16)f1.y; r[6]=(_Float16)f1.z; r[7]=(_Float16)f1.w;
    r[8]=(_Float16)f2.x; r[9]=(_Float16)f2.y; r[10]=(_Float16)f2.z; r[11]=(_Float16)f2.w;
    r[12]=(_Float16)f3.x; r[13]=(_Float16)f3.y; r[14]=(_Float16)f3.z; r[15]=(_Float16)f3.w;
    return r;
}

// Same striping, but source already f16 and contiguous (16B aligned chunks).
__device__ inline v16h frag_from_f16(const _Float16* base) {
    const v8h* p = (const v8h*)base;
    v8h lo = p[0];             // k+0..7
    v8h hi = p[2];             // k+16..23
    v16h r;
#pragma unroll
    for (int h = 0; h < 8; ++h) { r[h] = lo[h]; r[h + 8] = hi[h]; }
    return r;
}

// ---------------------------------------------------------------------------
// Kernel 1: Wh = x @ W1^T + b1  (f32 in, f32 WMMA-accum, store f16 transposed)
//   Wh_t[b][e][j] = Wh[b][j][e]   (j contiguous -> B-fragments of GEMM2 and
//   the e_l/e_r pass are fully coalesced)
// One wave -> 16 rows x 64 cols (4 n-tiles, A-fragment reused 4x).
// waves = (NROW/16) * (D/64) = 32768 ; blocks of 8 waves -> 4096 blocks.
// ---------------------------------------------------------------------------
__global__ __launch_bounds__(256) void gat_gemm1(
    const float* __restrict__ x, const float* __restrict__ W1,
    const float* __restrict__ b1, _Float16* __restrict__ Wh_t) {

    const int lane = threadIdx.x & 31;
    const int wid  = threadIdx.x >> 5;
    const int wg   = blockIdx.x * 8 + wid;
    const int rowTile = wg >> 2;              // 0..8191
    const int equad   = wg & 3;               // 0..3
    const int row0 = rowTile * 16;
    const int e0   = equad * 64;

    const int mm   = lane & 15;
    const int half = lane >> 4;
    const int klo  = half * 8;

    v8f acc0 = {}, acc1 = {}, acc2 = {}, acc3 = {};

#pragma unroll 2
    for (int kt = 0; kt < 8; ++kt) {
        const int k0 = kt * 32;
        v16h a = frag_from_f32(x + (row0 + mm) * Dc + k0 + klo);
        const int n = e0 + mm;
        v16h bf0 = frag_from_f32(W1 + (n +  0) * Dc + k0 + klo);
        v16h bf1 = frag_from_f32(W1 + (n + 16) * Dc + k0 + klo);
        v16h bf2 = frag_from_f32(W1 + (n + 32) * Dc + k0 + klo);
        v16h bf3 = frag_from_f32(W1 + (n + 48) * Dc + k0 + klo);
        acc0 = __builtin_amdgcn_wmma_f32_16x16x32_f16(false, a, false, bf0, (short)0, acc0, false, false);
        acc1 = __builtin_amdgcn_wmma_f32_16x16x32_f16(false, a, false, bf1, (short)0, acc1, false, false);
        acc2 = __builtin_amdgcn_wmma_f32_16x16x32_f16(false, a, false, bf2, (short)0, acc2, false, false);
        acc3 = __builtin_amdgcn_wmma_f32_16x16x32_f16(false, a, false, bf3, (short)0, acc3, false, false);
    }

    // Epilogue: add bias, store transposed f16. C/D layout: vgpr r ->
    // M = r + 8*half, N = mm. Lane stores 8 consecutive j's (16B aligned).
    const int bidx = row0 >> 8;               // batch (16 | 256 rows)
    const int jbase = (row0 & 255) + 8 * half;
    v8f accs[4] = {acc0, acc1, acc2, acc3};
#pragma unroll
    for (int nt = 0; nt < 4; ++nt) {
        const int n = e0 + nt * 16 + mm;
        const float bias = b1[n];
        v8h outp;
#pragma unroll
        for (int r = 0; r < 8; ++r)
            outp[r] = (_Float16)(accs[nt][r] + bias);
        *(v8h*)(Wh_t + (bidx * Dc + n) * Sc + jbase) = outp;
    }
}

// ---------------------------------------------------------------------------
// Kernel 2: e_l/e_r = Wh @ a_l / a_r, via transposed Wh (coalesced rows).
// 1 block per batch, 1 thread per token j.
// ---------------------------------------------------------------------------
__global__ __launch_bounds__(256) void gat_attn_vec(
    const _Float16* __restrict__ Wh_t, const float* __restrict__ a_w,
    float* __restrict__ e_l, float* __restrict__ e_r) {

    const int b = blockIdx.x;
    const int j = threadIdx.x;
    const _Float16* base = Wh_t + b * Dc * Sc + j;
    float el = 0.f, er = 0.f;
#pragma unroll 4
    for (int e = 0; e < Dc; ++e) {
        const float v = (float)base[e * Sc];
        el += v * a_w[e];
        er += v * a_w[Dc + e];
    }
    e_l[b * Sc + j] = el;
    e_r[b * Sc + j] = er;
}

// ---------------------------------------------------------------------------
// Kernel 3: fused attention softmax + aggregation + ELU.
// One block (8 waves) per (batch, 16-row i-tile): 512*16 = 8192 blocks.
//   s[i][j] = leaky(e_l[i]+e_r[j]+a_b) + pos[b,i,j]   -> LDS (padded +4)
//   row softmax (shfl reductions), p = exp(s-max), rinv = 1/sum
//   agg = (p*rinv) @ Wh_b  via WMMA, wave w owns n-tiles {w, w+8}
// ---------------------------------------------------------------------------
#define SPAD 260   // 256 + 4: 4*m bank skew -> conflict-free A-frag LDS reads

__global__ __launch_bounds__(256) void gat_attn_agg(
    const float* __restrict__ pos, const _Float16* __restrict__ Wh_t,
    const float* __restrict__ e_l, const float* __restrict__ e_r,
    const float* __restrict__ a_b, float* __restrict__ out) {

    __shared__ float sbuf[16][SPAD];
    __shared__ float rinvLds[16];

    const int b  = blockIdx.x >> 4;
    const int i0 = (blockIdx.x & 15) * 16;
    const int t  = threadIdx.x;
    const int lane = t & 31;
    const int wid  = t >> 5;

    // Phase 1: logits. thread t covers column j = t for all 16 rows.
    {
        const float ab = a_b[0];
        const float erj = e_r[b * Sc + t] + ab;
        const float* prow = pos + ((b * Sc + i0) * Sc + t);
#pragma unroll
        for (int i = 0; i < 16; ++i) {
            float s = e_l[b * Sc + i0 + i] + erj;
            s = (s >= 0.f) ? s : NEG_SLOPE * s;
            sbuf[i][t] = s + prow[i * Sc];
        }
    }
    __syncthreads();

    // Phase 2: row softmax stats. wave w handles rows 2w, 2w+1.
#pragma unroll
    for (int rr = 0; rr < 2; ++rr) {
        const int row = 2 * wid + rr;
        float m = -3.4e38f;
#pragma unroll
        for (int q = 0; q < 8; ++q) m = fmaxf(m, sbuf[row][lane + 32 * q]);
#pragma unroll
        for (int off = 16; off; off >>= 1) m = fmaxf(m, __shfl_xor(m, off, 32));
        float sum = 0.f;
#pragma unroll
        for (int q = 0; q < 8; ++q) {
            const float p = __expf(sbuf[row][lane + 32 * q] - m);
            sbuf[row][lane + 32 * q] = p;
            sum += p;
        }
#pragma unroll
        for (int off = 16; off; off >>= 1) sum += __shfl_xor(sum, off, 32);
        if (lane == 0) rinvLds[row] = 1.f / sum;
    }
    __syncthreads();

    // Phase 3: agg = w @ Wh_b, WMMA. wave wid owns n0 = {wid,wid+8}*16.
    const int mm   = lane & 15;
    const int half = lane >> 4;
    const int klo  = half * 8;
    const float ri = rinvLds[mm];

    v8f acc0 = {}, acc1 = {};
    const _Float16* whb = Wh_t + b * Dc * Sc;
#pragma unroll 2
    for (int kt = 0; kt < 8; ++kt) {
        const int k0 = kt * 32;
        v16h a;
#pragma unroll
        for (int h = 0; h < 8; ++h) {
            a[h]     = (_Float16)(sbuf[mm][k0 + klo + h]      * ri);
            a[h + 8] = (_Float16)(sbuf[mm][k0 + klo + 16 + h] * ri);
        }
        v16h bf0 = frag_from_f16(whb + ((wid +  0) * 16 + mm) * Sc + k0 + klo);
        v16h bf1 = frag_from_f16(whb + ((wid +  8) * 16 + mm) * Sc + k0 + klo);
        acc0 = __builtin_amdgcn_wmma_f32_16x16x32_f16(false, a, false, bf0, (short)0, acc0, false, false);
        acc1 = __builtin_amdgcn_wmma_f32_16x16x32_f16(false, a, false, bf1, (short)0, acc1, false, false);
    }

    // Epilogue: ELU, store f32. M = r + 8*half, N = mm.
#pragma unroll
    for (int nt = 0; nt < 2; ++nt) {
        const v8f acc = nt ? acc1 : acc0;
        const int n = (wid + nt * 8) * 16 + mm;
#pragma unroll
        for (int r = 0; r < 8; ++r) {
            const int i = i0 + r + 8 * half;
            float v = acc[r];
            v = (v >= 0.f) ? v : (__expf(v) - 1.f);
            out[(b * Sc + i) * Dc + n] = v;
        }
    }
}

// ---------------------------------------------------------------------------
extern "C" void kernel_launch(void* const* d_in, const int* in_sizes, int n_in,
                              void* d_out, int out_size, void* d_ws, size_t ws_size,
                              hipStream_t stream) {
    (void)in_sizes; (void)n_in; (void)out_size; (void)ws_size;
    const float* x   = (const float*)d_in[0];
    const float* pos = (const float*)d_in[1];
    const float* W1  = (const float*)d_in[2];
    const float* b1  = (const float*)d_in[3];
    const float* a_w = (const float*)d_in[4];
    const float* a_b = (const float*)d_in[5];
    float* out = (float*)d_out;

    char* ws = (char*)d_ws;
    _Float16* Wh_t = (_Float16*)ws;                             // 67,108,864 B
    float* e_l = (float*)(ws + (size_t)NROW * Dc * 2);          //    524,288 B
    float* e_r = e_l + NROW;                                    //    524,288 B

    gat_gemm1<<<dim3((NROW / 16) * (Dc / 64) / 8), dim3(256), 0, stream>>>(x, W1, b1, Wh_t);
    gat_attn_vec<<<dim3(Bc), dim3(Sc), 0, stream>>>(Wh_t, a_w, e_l, e_r);
    gat_attn_agg<<<dim3(Bc * (Sc / 16)), dim3(256), 0, stream>>>(pos, Wh_t, e_l, e_r, a_b, out);
}
